// InterpreterWithRegistersAndKbit_31318901522621
// MI455X (gfx1250) — compile-verified, matrix operand order checked by reading, not run on hardware
//
#include <hip/hip_runtime.h>
#include <cstdint>
#include <cstddef>

// ---------------------------------------------------------------------------
// InterpreterWithRegisters scan for MI455X (gfx1250, wave32, WMMA).
//
// Sizes: B=256, LINES=64, NREG=64, D=1024, L=64, K=16, F=L*K=1024.
// Total GEMM math ~103 GFLOP; weights (12MB) + R state (64MB) are L2-resident
// (192MB L2), so the scan is latency bound -> f16 WMMA for all three
// 256x1024x1024 GEMMs per step, R updated in place in d_out.
// A-blocks are staged into LDS once per workgroup with the CDNA5 async
// global->LDS copies (ASYNCcnt) so only B fragments stream from L2 in the
// WMMA inner loop.
// ---------------------------------------------------------------------------

static constexpr int BATCH = 256;
static constexpr int NLINES = 64;
static constexpr int NREGS = 64;
static constexpr int DDIM = 1024;   // register width D
static constexpr int FDIM = 1024;   // L*K
static constexpr float EPSV = 1e-5f;
static constexpr float SCALEV = 0.03125f;  // 1/sqrt(1024)

typedef __attribute__((ext_vector_type(16))) _Float16 v16h;
typedef __attribute__((ext_vector_type(8)))  float    v8f;
typedef __attribute__((ext_vector_type(4)))  unsigned int u32x4;

union FragH {
  u32x4 u4[2];
  v16h  v;
};

// Async global->LDS 16B copy (CDNA5 GLOBAL_LOAD_ASYNC_TO_LDS_B128, ASYNCcnt).
__device__ __forceinline__ void async_copy_b128(uint32_t lds_off, const void* gaddr) {
  asm volatile("global_load_async_to_lds_b128 %0, %1, off"
               :: "v"(lds_off), "v"((uint64_t)(uintptr_t)gaddr)
               : "memory");
}
__device__ __forceinline__ void async_wait0() {
  asm volatile("s_wait_asynccnt 0x0" ::: "memory");
}

// ---------------------------------------------------------------------------
// One-time: transpose 1024x1024 f32 (K-major) -> f16 [N][K] so WMMA B
// fragments are contiguous 16B loads per lane. Tiled via LDS, coalesced.
// ---------------------------------------------------------------------------
__global__ void transpose_convert_kernel(const float* __restrict__ src,
                                         _Float16* __restrict__ dst) {
  __shared__ float tile[32][33];
  const int tx = threadIdx.x & 31;
  const int ty = threadIdx.x >> 5;  // 0..7
  const int c0 = blockIdx.x * 32;   // src col base (N)
  const int r0 = blockIdx.y * 32;   // src row base (K)
  for (int i = 0; i < 4; ++i) {
    int r = ty + i * 8;
    tile[r][tx] = src[(size_t)(r0 + r) * 1024 + (c0 + tx)];
  }
  __syncthreads();
  for (int i = 0; i < 4; ++i) {
    int r = ty + i * 8;  // local N index
    dst[(size_t)(c0 + r) * 1024 + (r0 + tx)] = (_Float16)tile[tx][r];
  }
}

// ---------------------------------------------------------------------------
// Step phase 1: attention read (2 queries over 64 registers) + LN1/LN2,
// emitting f16 GEMM inputs xin/yin (one row per batch element).
// One workgroup (256 threads) per batch element.
// ---------------------------------------------------------------------------
__global__ void attn_read_kernel(const float* __restrict__ R,
                                 const float* __restrict__ q_read,
                                 const float* __restrict__ ln1_g,
                                 const float* __restrict__ ln1_b,
                                 const float* __restrict__ ln2_g,
                                 const float* __restrict__ ln2_b,
                                 _Float16* __restrict__ xin,
                                 _Float16* __restrict__ yin,
                                 int t) {
  __shared__ float part[256];
  __shared__ float attn[2][64];
  __shared__ float ops[2][1024];
  __shared__ float red[256];

  const int b = blockIdx.x;
  const int tid = threadIdx.x;

  // logits: 128 dot products (2 queries x 64 regs), 2 threads per dot
  {
    const int dot = tid & 127;
    const int pt = tid >> 7;
    const int q = dot >> 6;
    const int r = dot & 63;
    const float* qp = q_read + ((size_t)(b * NLINES + t) * 2 + q) * DDIM;
    const float* rp = R + (size_t)(b * NREGS + r) * DDIM;
    float s = 0.f;
    for (int d = pt * 512; d < pt * 512 + 512; ++d) s += qp[d] * rp[d];
    part[tid] = s;
  }
  __syncthreads();
  if (tid < 128) {
    const int q = tid >> 6, r = tid & 63;
    attn[q][r] = (part[tid] + part[tid + 128]) * SCALEV;
  }
  __syncthreads();
  if (tid < 2) {  // softmax over 64 registers for query q=tid
    float mx = -1e30f;
    for (int r = 0; r < 64; ++r) mx = fmaxf(mx, attn[tid][r]);
    float sum = 0.f;
    for (int r = 0; r < 64; ++r) {
      float e = __expf(attn[tid][r] - mx);
      attn[tid][r] = e;
      sum += e;
    }
    float inv = 1.f / sum;
    for (int r = 0; r < 64; ++r) attn[tid][r] *= inv;
  }
  __syncthreads();

  // op_s[q][d] = sum_r attn[q][r] * R[b][r][d]; one pass over R[b]
  float a0[4] = {0.f, 0.f, 0.f, 0.f}, a1[4] = {0.f, 0.f, 0.f, 0.f};
  for (int r = 0; r < 64; ++r) {
    const float w0 = attn[0][r], w1v = attn[1][r];
    const float* rp = R + (size_t)(b * NREGS + r) * DDIM + tid;
    for (int i = 0; i < 4; ++i) {
      float v = rp[i * 256];
      a0[i] += w0 * v;
      a1[i] += w1v * v;
    }
  }
  for (int i = 0; i < 4; ++i) {
    ops[0][tid + i * 256] = a0[i];
    ops[1][tid + i * 256] = a1[i];
  }
  __syncthreads();

  // LayerNorm each operand, store as f16
  for (int q = 0; q < 2; ++q) {
    float s = 0.f, ss = 0.f;
    for (int i = 0; i < 4; ++i) {
      float v = ops[q][tid + i * 256];
      s += v;
      ss += v * v;
    }
    red[tid] = s;
    __syncthreads();
    for (int o = 128; o > 0; o >>= 1) {
      if (tid < o) red[tid] += red[tid + o];
      __syncthreads();
    }
    const float mean = red[0] * (1.f / 1024.f);
    __syncthreads();
    red[tid] = ss;
    __syncthreads();
    for (int o = 128; o > 0; o >>= 1) {
      if (tid < o) red[tid] += red[tid + o];
      __syncthreads();
    }
    const float var = red[0] * (1.f / 1024.f) - mean * mean;
    __syncthreads();
    const float inv = rsqrtf(var + EPSV);
    const float* g = q ? ln2_g : ln1_g;
    const float* bb = q ? ln2_b : ln1_b;
    _Float16* outp = q ? yin : xin;
    for (int i = 0; i < 4; ++i) {
      int d = tid + i * 256;
      float v = (ops[q][d] - mean) * inv * g[d] + bb[d];
      outp[(size_t)b * DDIM + d] = (_Float16)v;
    }
    __syncthreads();
  }
}

// ---------------------------------------------------------------------------
// Step phase 2: C = A(256x1024,f16) @ Wt^T(1024x1024,f16) + bias, then
// softmax over each aligned group of 16 output columns. blockIdx.z picks the
// (xin,w1)->x or (yin,w2)->y problem. 512 threads = 16 waves; one workgroup
// covers a 16-row M tile across all N=1024. The A block (32KB) is staged in
// LDS via async copies; WMMA A-frags come from ds_load, B streams from L2.
// ---------------------------------------------------------------------------
__global__ void gemm_xy_kernel(const _Float16* __restrict__ xin,
                               const _Float16* __restrict__ yin,
                               const _Float16* __restrict__ Wt1,
                               const _Float16* __restrict__ Wt2,
                               const float* __restrict__ b1,
                               const float* __restrict__ b2,
                               float* __restrict__ xo,
                               float* __restrict__ yo) {
  __shared__ _Float16 Asm[16 * 1024];  // 32KB staged A block

  const int z = blockIdx.z;
  const _Float16* A = z ? yin : xin;
  const _Float16* Wt = z ? Wt2 : Wt1;   // [N][K] layout
  const float* bias = z ? b2 : b1;
  float* Out = z ? yo : xo;

  const int wave = threadIdx.x >> 5;   // 0..15
  const int lane = threadIdx.x & 31;
  const int lm = lane & 15;
  const int half = lane >> 4;

  const int m_base = blockIdx.x * 16;
  const int n_wave = wave * 64;

  // stage A rows m_base..m_base+15 (contiguous 32KB) into LDS asynchronously
  {
    const uint32_t lds_base = (uint32_t)(uintptr_t)(void*)Asm;
    const _Float16* gsrc = A + (size_t)m_base * DDIM;
    for (int c = 0; c < 4; ++c) {
      int elem = (threadIdx.x + c * 512) * 8;  // 8 halfs = 16B per chunk
      async_copy_b128(lds_base + (uint32_t)elem * 2, gsrc + elem);
    }
    async_wait0();
    __syncthreads();
  }

  const _Float16* ArowL = Asm + lm * 1024;

  v8f acc[4] = {};
  for (int kk = 0; kk < DDIM; kk += 32) {
    FragH a;
    a.u4[0] = *reinterpret_cast<const u32x4*>(ArowL + kk + half * 8);
    a.u4[1] = *reinterpret_cast<const u32x4*>(ArowL + kk + 16 + half * 8);
#pragma unroll
    for (int tt = 0; tt < 4; ++tt) {
      const _Float16* Brow = Wt + (size_t)(n_wave + tt * 16 + lm) * DDIM;
      FragH bf;
      bf.u4[0] = *reinterpret_cast<const u32x4*>(Brow + kk + half * 8);
      bf.u4[1] = *reinterpret_cast<const u32x4*>(Brow + kk + 16 + half * 8);
      acc[tt] = __builtin_amdgcn_wmma_f32_16x16x32_f16(
          false, a.v, false, bf.v, (short)0, acc[tt], false, false);
    }
  }

  // epilogue: +bias, softmax over the 16 columns of each 16x16 tile.
  // C layout: lane holds column n=lm, VGPR j holds row j+8*half.
#pragma unroll
  for (int tt = 0; tt < 4; ++tt) {
    const int n = n_wave + tt * 16 + lm;
    const float bv = bias[n];
#pragma unroll
    for (int j = 0; j < 8; ++j) {
      float v = acc[tt][j] + bv;
      float mx = v;
      for (int s = 1; s < 16; s <<= 1) mx = fmaxf(mx, __shfl_xor(mx, s, 32));
      float e = __expf(v - mx);
      float sm = e;
      for (int s = 1; s < 16; s <<= 1) sm += __shfl_xor(sm, s, 32);
      Out[(size_t)(m_base + j + 8 * half) * FDIM + n] = e / sm;
    }
  }
}

// ---------------------------------------------------------------------------
// Step phase 3: modular digit mixing. out_add[m]=sum_i x_i*y_{(m-i)%16},
// out_sub[m]=sum_i x_i*y_{(i-m)%16}; mix = p0*add + p1*sub, stored f16.
// One workgroup per batch element; thread quartets share a line.
// ---------------------------------------------------------------------------
__global__ void digit_mix_kernel(const float* __restrict__ xo,
                                 const float* __restrict__ yo,
                                 const float* __restrict__ opcode_probs,
                                 _Float16* __restrict__ mix,
                                 int t) {
  const int b = blockIdx.x;
  const int tid = threadIdx.x;
  const int l = tid >> 2;
  const int s4 = tid & 3;
  const float p0 = opcode_probs[(size_t)(b * NLINES + t) * 2 + 0];
  const float p1 = opcode_probs[(size_t)(b * NLINES + t) * 2 + 1];
  float xl[16], yl[16];
  const float* xp = xo + (size_t)b * FDIM + l * 16;
  const float* yp = yo + (size_t)b * FDIM + l * 16;
  for (int i = 0; i < 16; ++i) {
    xl[i] = xp[i];
    yl[i] = yp[i];
  }
  for (int mm = 0; mm < 4; ++mm) {
    const int m = s4 * 4 + mm;
    float add = 0.f, sub = 0.f;
#pragma unroll
    for (int i = 0; i < 16; ++i) {
      add += xl[i] * yl[(m - i) & 15];
      sub += xl[i] * yl[(i - m) & 15];
    }
    mix[(size_t)b * FDIM + l * 16 + m] = (_Float16)(p0 * add + p1 * sub);
  }
}

// ---------------------------------------------------------------------------
// Step phase 4: value = LN(mix @ w_res + b_res). One workgroup (8 waves)
// covers a full 16-row M tile and all 1024 output columns so the LayerNorm
// row reduction stays inside the block. Per wave: 16Mx128N, 8 accumulators.
// A block staged in LDS via async copies, as in gemm_xy.
// ---------------------------------------------------------------------------
__global__ void gemm_res_kernel(const _Float16* __restrict__ mix,
                                const _Float16* __restrict__ WresT,
                                const float* __restrict__ b_res,
                                const float* __restrict__ lno_g,
                                const float* __restrict__ lno_b,
                                float* __restrict__ value) {
  __shared__ _Float16 Asm[16 * 1024];  // 32KB staged A block
  __shared__ float sumL[8 * 16];
  __shared__ float sqL[8 * 16];
  __shared__ float meanL[16];
  __shared__ float invL[16];

  const int wave = threadIdx.x >> 5;
  const int lane = threadIdx.x & 31;
  const int lm = lane & 15;
  const int half = lane >> 4;
  const int m_base = blockIdx.x * 16;
  const int n_wave = wave * 128;

  {
    const uint32_t lds_base = (uint32_t)(uintptr_t)(void*)Asm;
    const _Float16* gsrc = mix + (size_t)m_base * FDIM;
    for (int c = 0; c < 8; ++c) {
      int elem = (threadIdx.x + c * 256) * 8;
      async_copy_b128(lds_base + (uint32_t)elem * 2, gsrc + elem);
    }
    async_wait0();
    __syncthreads();
  }

  const _Float16* ArowL = Asm + lm * 1024;

  v8f acc[8] = {};
  for (int kk = 0; kk < FDIM; kk += 32) {
    FragH a;
    a.u4[0] = *reinterpret_cast<const u32x4*>(ArowL + kk + half * 8);
    a.u4[1] = *reinterpret_cast<const u32x4*>(ArowL + kk + 16 + half * 8);
#pragma unroll
    for (int tt = 0; tt < 8; ++tt) {
      const _Float16* Brow = WresT + (size_t)(n_wave + tt * 16 + lm) * FDIM;
      FragH bf;
      bf.u4[0] = *reinterpret_cast<const u32x4*>(Brow + kk + half * 8);
      bf.u4[1] = *reinterpret_cast<const u32x4*>(Brow + kk + 16 + half * 8);
      acc[tt] = __builtin_amdgcn_wmma_f32_16x16x32_f16(
          false, a.v, false, bf.v, (short)0, acc[tt], false, false);
    }
  }

  // +bias before LN statistics
#pragma unroll
  for (int tt = 0; tt < 8; ++tt) {
    const float bv = b_res[n_wave + tt * 16 + lm];
#pragma unroll
    for (int j = 0; j < 8; ++j) acc[tt][j] += bv;
  }

  // per-row partial sums over this wave's 128 columns
#pragma unroll
  for (int j = 0; j < 8; ++j) {
    float s = 0.f, q = 0.f;
#pragma unroll
    for (int tt = 0; tt < 8; ++tt) {
      float v = acc[tt][j];
      s += v;
      q += v * v;
    }
    for (int st = 1; st < 16; st <<= 1) {
      s += __shfl_xor(s, st, 32);
      q += __shfl_xor(q, st, 32);
    }
    if (lm == 0) {
      sumL[wave * 16 + j + 8 * half] = s;
      sqL[wave * 16 + j + 8 * half] = q;
    }
  }
  __syncthreads();
  if (threadIdx.x < 16) {
    const int row = threadIdx.x;
    float S = 0.f, Q = 0.f;
    for (int w = 0; w < 8; ++w) {
      S += sumL[w * 16 + row];
      Q += sqL[w * 16 + row];
    }
    const float mean = S * (1.f / 1024.f);
    const float var = Q * (1.f / 1024.f) - mean * mean;
    meanL[row] = mean;
    invL[row] = rsqrtf(var + EPSV);
  }
  __syncthreads();

#pragma unroll
  for (int tt = 0; tt < 8; ++tt) {
    const int n = n_wave + tt * 16 + lm;
    const float g = lno_g[n], bb = lno_b[n];
#pragma unroll
    for (int j = 0; j < 8; ++j) {
      const int row = j + 8 * half;
      value[(size_t)(m_base + row) * DDIM + n] =
          (acc[tt][j] - meanL[row]) * invL[row] * g + bb;
    }
  }
}

// ---------------------------------------------------------------------------
// Step phase 5: gated attention write-back. w = softmax(kw.R/32) over 64
// registers; R = R*(1-g*w) + g*w*value. One workgroup per batch element.
// ---------------------------------------------------------------------------
__global__ void write_kernel(float* __restrict__ R,
                             const float* __restrict__ k_write,
                             const float* __restrict__ gate,
                             const float* __restrict__ value,
                             int t) {
  __shared__ float part[256];
  __shared__ float gw[64];
  const int b = blockIdx.x;
  const int tid = threadIdx.x;
  {
    const int r = tid & 63;
    const int pt = tid >> 6;
    const float* kp = k_write + (size_t)(b * NLINES + t) * DDIM;
    const float* rp = R + (size_t)(b * NREGS + r) * DDIM;
    float s = 0.f;
    for (int d = pt * 256; d < pt * 256 + 256; ++d) s += kp[d] * rp[d];
    part[tid] = s;
  }
  __syncthreads();
  if (tid == 0) {
    const float g = gate[b * NLINES + t];
    float lg[64];
    float mx = -1e30f;
    for (int r = 0; r < 64; ++r) {
      float v = (part[r] + part[64 + r] + part[128 + r] + part[192 + r]) * SCALEV;
      lg[r] = v;
      mx = fmaxf(mx, v);
    }
    float sum = 0.f;
    for (int r = 0; r < 64; ++r) {
      lg[r] = __expf(lg[r] - mx);
      sum += lg[r];
    }
    const float inv = g / sum;
    for (int r = 0; r < 64; ++r) gw[r] = lg[r] * inv;
  }
  __syncthreads();
  float val[4];
  for (int i = 0; i < 4; ++i) val[i] = value[(size_t)b * DDIM + tid + i * 256];
  for (int r = 0; r < 64; ++r) {
    const float grv = gw[r];
    float* rp = R + (size_t)(b * NREGS + r) * DDIM + tid;
    for (int i = 0; i < 4; ++i) {
      float old = rp[i * 256];
      rp[i * 256] = old * (1.f - grv) + grv * val[i];
    }
  }
}

// ---------------------------------------------------------------------------
// Host driver: convert weights once, copy R init into d_out, then 64 steps
// of 5 kernels each on `stream` (graph-capture safe, deterministic).
// ---------------------------------------------------------------------------
extern "C" void kernel_launch(void* const* d_in, const int* in_sizes, int n_in,
                              void* d_out, int out_size, void* d_ws, size_t ws_size,
                              hipStream_t stream) {
  (void)in_sizes; (void)n_in; (void)out_size; (void)ws_size;

  const float* opcode_probs = (const float*)d_in[0];
  const float* registers    = (const float*)d_in[1];
  const float* k_write      = (const float*)d_in[2];
  const float* q_read       = (const float*)d_in[3];
  /* d_in[4] = cond_distribution: unused by the reference */
  const float* gate  = (const float*)d_in[5];
  const float* ln1_g = (const float*)d_in[6];
  const float* ln1_b = (const float*)d_in[7];
  const float* w1    = (const float*)d_in[8];
  const float* b1    = (const float*)d_in[9];
  const float* ln2_g = (const float*)d_in[10];
  const float* ln2_b = (const float*)d_in[11];
  const float* w2    = (const float*)d_in[12];
  const float* b2    = (const float*)d_in[13];
  const float* w_res = (const float*)d_in[14];
  const float* b_res = (const float*)d_in[15];
  const float* lno_g = (const float*)d_in[16];
  const float* lno_b = (const float*)d_in[17];

  float* R = (float*)d_out;  // R state lives in the output buffer

  char* ws = (char*)d_ws;
  size_t off = 0;
  auto take = [&](size_t bytes) -> void* {
    void* p = ws + off;
    off += (bytes + 255) & ~(size_t)255;
    return p;
  };
  _Float16* Wt1 = (_Float16*)take((size_t)1024 * 1024 * 2);   // w1^T  f16 [F][D]
  _Float16* Wt2 = (_Float16*)take((size_t)1024 * 1024 * 2);   // w2^T  f16 [F][D]
  _Float16* WtR = (_Float16*)take((size_t)1024 * 1024 * 2);   // w_res^T f16 [D][F]
  _Float16* xin = (_Float16*)take((size_t)BATCH * DDIM * 2);
  _Float16* yin = (_Float16*)take((size_t)BATCH * DDIM * 2);
  float* xo     = (float*)take((size_t)BATCH * FDIM * 4);
  float* yo     = (float*)take((size_t)BATCH * FDIM * 4);
  _Float16* mix = (_Float16*)take((size_t)BATCH * FDIM * 2);
  float* value  = (float*)take((size_t)BATCH * DDIM * 4);

  dim3 tgrid(32, 32);
  transpose_convert_kernel<<<tgrid, 256, 0, stream>>>(w1, Wt1);
  transpose_convert_kernel<<<tgrid, 256, 0, stream>>>(w2, Wt2);
  transpose_convert_kernel<<<tgrid, 256, 0, stream>>>(w_res, WtR);
  hipMemcpyAsync(R, registers, (size_t)BATCH * NREGS * DDIM * sizeof(float),
                 hipMemcpyDeviceToDevice, stream);

  for (int t = 0; t < NLINES; ++t) {
    attn_read_kernel<<<BATCH, 256, 0, stream>>>(R, q_read, ln1_g, ln1_b,
                                                ln2_g, ln2_b, xin, yin, t);
    gemm_xy_kernel<<<dim3(16, 1, 2), 512, 0, stream>>>(xin, yin, Wt1, Wt2,
                                                       b1, b2, xo, yo);
    digit_mix_kernel<<<BATCH, 256, 0, stream>>>(xo, yo, opcode_probs, mix, t);
    gemm_res_kernel<<<16, 256, 0, stream>>>(mix, WtR, b_res, lno_g, lno_b, value);
    write_kernel<<<BATCH, 256, 0, stream>>>(R, k_write, gate, value, t);
  }
}
